// CoupledModel_24670292148952
// MI455X (gfx1250) — matrix operand order repaired and checked by reading, not compiled, not run on hardware
//
#include <hip/hip_runtime.h>
#include <hip/hip_bf16.h>

typedef __attribute__((ext_vector_type(16))) _Float16 v16h;
typedef __attribute__((ext_vector_type(8)))  float    v8f;
typedef __attribute__((ext_vector_type(4)))  int      vint4;

#define NUM_NETS 24
#define WIDTH 64
#define ITERS 4
#define SAMPLES_PER_BLOCK (8 * 16 * ITERS)   // 8 waves * 16 rows * ITERS

#if __has_builtin(__builtin_amdgcn_global_load_async_to_lds_b128) && \
    __has_builtin(__builtin_amdgcn_s_wait_asynccnt)
#define ASYNC_STAGE 1
#else
#define ASYNC_STAGE 0
#endif

union V16H {
    v16h     v;
    uint32_t u[8];
    _Float16 f[16];
};

// Branch-free tanh: hardware v_tanh_f32 when available, else exp2+rcp.
__device__ __forceinline__ float fast_tanh(float x) {
#if __has_builtin(__builtin_amdgcn_tanhf)
    return __builtin_amdgcn_tanhf(x);
#else
    float xc = fminf(fmaxf(x, -10.0f), 10.0f);
    float e  = __builtin_amdgcn_exp2f(xc * 2.8853900817779268f);
    return (e - 1.0f) * __builtin_amdgcn_rcpf(e + 1.0f);
#endif
}

// Pack two f32 -> one dword of packed f16 with a single v_cvt_pk_f16_f32.
__device__ __forceinline__ uint32_t pack_f16(float a, float b) {
    return __builtin_bit_cast(uint32_t, __builtin_amdgcn_cvt_pkrtz(a, b));
}

// ---------------------------------------------------------------------------
// Pass 1: for each network k, sub[n,k] = W3[k] . tanh(W2[k] tanh(W1[k] x + b1) + b2)
// Layer-2 (64x64 GEMM, the FLOP hot spot) runs on v_wmma_f32_16x16x32_f16.
// ---------------------------------------------------------------------------
__global__ __launch_bounds__(256)
void mlp_sub_kernel(const float* __restrict__ x,
                    const float* __restrict__ W1,
                    const float* __restrict__ b1,
                    const float* __restrict__ W2,
                    const float* __restrict__ b2,
                    const float* __restrict__ W3,
                    float* __restrict__ sub,
                    int N)
{
    __shared__ _Float16 sW2[WIDTH * WIDTH];   // [v][w] row-major, f16
#if ASYNC_STAGE
    __shared__ float    sW2f32[WIDTH * WIDTH];  // async-staged f32 copy
#endif
    __shared__ float    sW1[WIDTH * 2];
    __shared__ float    sb1[WIDTH];
    __shared__ float    sb2[WIDTH];
    __shared__ float    sW3[WIDTH];

    const int k   = blockIdx.y;
    const int tid = threadIdx.x;

    // Small weights: plain staging.
    if (tid < 2 * WIDTH) sW1[tid] = W1[k * 2 * WIDTH + tid];
    if (tid < WIDTH) {
        sb1[tid] = b1[k * WIDTH + tid];
        sb2[tid] = b2[k * WIDTH + tid];
        sW3[tid] = W3[k * WIDTH + tid];
    }

#if ASYNC_STAGE
    // W2 (16 KB f32): async global->LDS DMA (no VGPR round-trip), then one
    // LDS-local f32->f16 conversion pass.
    {
        const float* gW2 = W2 + k * WIDTH * WIDTH;
#pragma unroll
        for (int tr = 0; tr < 4; ++tr) {
            const int e = (tid + tr * 256) * 4;    // 4 floats per b128 transfer
            __builtin_amdgcn_global_load_async_to_lds_b128(
                (__attribute__((address_space(1))) vint4*)(gW2 + e),
                (__attribute__((address_space(3))) vint4*)&sW2f32[e], 0, 0);
        }
        __builtin_amdgcn_s_wait_asynccnt(0);
        __syncthreads();
        const int base = tid * 16;                 // 16 contiguous elems/thread
#pragma unroll
        for (int e = 0; e < 16; e += 2)
            *reinterpret_cast<uint32_t*>(&sW2[base + e]) =
                pack_f16(sW2f32[base + e], sW2f32[base + e + 1]);
        __syncthreads();
    }
#else
    for (int i = tid; i < WIDTH * WIDTH; i += 256)
        sW2[i] = (_Float16)W2[k * WIDTH * WIDTH + i];
    __syncthreads();
#endif

    const int lane = tid & 31;
    const int wave = tid >> 5;
    const int h    = lane >> 4;    // half-wave index (0/1)
    const int nl   = lane & 15;    // lane within half

    // B fragments (weight-stationary in registers):
    // B[t][p] = 32(w) x 16(v) tile of W2^T, w in [32p,32p+32), v in [16t,16t+16).
    // 16-bit B layout: lane holds column v = nl; VGPR j packs K pair (j<4: K=2j+8h; j>=4: K=16+2(j-4)+8h).
    V16H B[4][2];
#pragma unroll
    for (int t = 0; t < 4; ++t)
#pragma unroll
        for (int p = 0; p < 2; ++p)
#pragma unroll
            for (int j = 0; j < 8; ++j) {
                const int w = p * 32 + 2 * (j & 3) + 8 * h + 16 * (j >> 2);
                const int v = t * 16 + nl;
                B[t][p].u[j] = *reinterpret_cast<const uint32_t*>(&sW2[v * WIDTH + w]);
            }

    const int blockBase = blockIdx.x * SAMPLES_PER_BLOCK + wave * (ITERS * 16);

    // unroll 2 (not 4): keeps live ranges short enough that the 64 B-fragment
    // VGPRs stay register-resident instead of being re-streamed from LDS.
#pragma unroll 2
    for (int it = 0; it < ITERS; ++it) {
        const int base = blockBase + it * 16;
        int n = base + nl;
        if (n >= N) n = N - 1;          // clamp: keeps EXEC all-ones for WMMA
        const float x0 = x[2 * n + 0];
        const float x1 = x[2 * n + 1];

        // A fragments: h1 tile 16(samples) x 64(w), computed straight into the
        // 16-bit A VGPR layout (j<4: K=2j+8h; j>=4: K=16+2(j-4)+8h).
        V16H A[2];
#pragma unroll
        for (int p = 0; p < 2; ++p)
#pragma unroll
            for (int j = 0; j < 8; ++j) {
                const int kk = (j < 4) ? (2 * j + 8 * h) : (16 + 2 * (j - 4) + 8 * h);
                const int w  = p * 32 + kk;
                const float a0 = fast_tanh(fmaf(x0, sW1[2 * w + 0], fmaf(x1, sW1[2 * w + 1], sb1[w])));
                const float a1 = fast_tanh(fmaf(x0, sW1[2 * w + 2], fmaf(x1, sW1[2 * w + 3], sb1[w + 1])));
                A[p].u[j] = pack_f16(a0, a1);
            }

        float acc[8];
#pragma unroll
        for (int r = 0; r < 8; ++r) acc[r] = 0.f;

#pragma unroll
        for (int t = 0; t < 4; ++t) {
            v8f c = {0.f, 0.f, 0.f, 0.f, 0.f, 0.f, 0.f, 0.f};
            c = __builtin_amdgcn_wmma_f32_16x16x32_f16(false, A[0].v, false, B[t][0].v,
                                                       (short)0, c, false, false);
            c = __builtin_amdgcn_wmma_f32_16x16x32_f16(false, A[1].v, false, B[t][1].v,
                                                       (short)0, c, false, false);
            // C layout: lane -> column v, VGPR r -> row (r + 8h).
            const int   v  = t * 16 + nl;
            const float bb = sb2[v];
            const float w3 = sW3[v];
#pragma unroll
            for (int r = 0; r < 8; ++r) {
                const float h2v = fast_tanh(c[r] + bb);
                acc[r] = fmaf(h2v, w3, acc[r]);
            }
        }

        // Butterfly-reduce the v dimension across the 16 lanes of each half-wave.
#pragma unroll
        for (int m = 1; m <= 8; m <<= 1)
#pragma unroll
            for (int r = 0; r < 8; ++r)
                acc[r] += __shfl_xor(acc[r], m, 32);

        if (nl < 8) {
            float val = acc[0];
#pragma unroll
            for (int r = 1; r < 8; ++r)
                if (nl == r) val = acc[r];
            const int row = base + 8 * h + nl;
            if (row < N) sub[row * NUM_NETS + k] = val;
        }
    }
}

// ---------------------------------------------------------------------------
// Pass 2: per-sample free-energy softmax mix over the 24 networks.
// Memory-bound (~14 MB total) -> trivial at 23.3 TB/s. Native exp/log
// (branch-free v_exp_f32/v_log_f32); exponent clamped <= 10, w in (0,1].
// ---------------------------------------------------------------------------
__global__ __launch_bounds__(256)
void mix_kernel(const float* __restrict__ x,
                const float* __restrict__ sub,
                float* __restrict__ out, int N)
{
    const int n = blockIdx.x * blockDim.x + threadIdx.x;
    if (n >= N) return;
    const float kb = 0.1f;
    const float T  = fmaxf(x[2 * n + 1], 0.1f);
    const float* srow = sub + n * NUM_NETS;

    float vals[12], logits[12];
    float s = 0.f;
#pragma unroll
    for (int i = 0; i < 12; ++i) {
        const float n1 = srow[2 * i + 0];
        const float n2 = srow[2 * i + 1];
        const float v  = n1 - T * (n2 * n2);
        vals[i] = v;
        const float e = fminf(-v / (kb * T), 10.0f);
        logits[i] = __expf(e);
        s += logits[i];
    }
    s += 1e-9f;
    const float inv_s = 1.0f / s;
    float wv = 0.f, wlp = 0.f;
#pragma unroll
    for (int i = 0; i < 12; ++i) {
        const float w = logits[i] * inv_s;
        wv  = fmaf(w, vals[i], wv);
        wlp = fmaf(w, __logf(w + 1e-9f), wlp);
    }
    out[n] = fmaf(kb * T, wlp, wv);
}

// ---------------------------------------------------------------------------
extern "C" void kernel_launch(void* const* d_in, const int* in_sizes, int n_in,
                              void* d_out, int out_size, void* d_ws, size_t ws_size,
                              hipStream_t stream)
{
    const float* x  = (const float*)d_in[0];
    const float* W1 = (const float*)d_in[1];
    const float* b1 = (const float*)d_in[2];
    const float* W2 = (const float*)d_in[3];
    const float* b2 = (const float*)d_in[4];
    const float* W3 = (const float*)d_in[5];

    const int N = in_sizes[0] / 2;          // x is (N,2)
    float* out = (float*)d_out;             // (N,1) flattened
    float* sub = out + N;                   // (N,24) flattened, second tuple output

    dim3 grid1((N + SAMPLES_PER_BLOCK - 1) / SAMPLES_PER_BLOCK, NUM_NETS);
    mlp_sub_kernel<<<grid1, 256, 0, stream>>>(x, W1, b1, W2, b2, W3, sub, N);
    mix_kernel<<<(N + 255) / 256, 256, 0, stream>>>(x, sub, out, N);
}
